// base_Model_40621800685765
// MI455X (gfx1250) — matrix operand order repaired and checked by reading, not compiled, not run on hardware
//
#include <hip/hip_runtime.h>
#include <math.h>
#include <stdint.h>

// ---------------- problem constants ----------------
#define Bsz 16
#define Lseq 512
#define Cin 38
#define Dm 512
#define Hh 8
#define Eh 64
#define DFFm 2048
#define NLm 3
#define DRESm 128
#define Mrows (Bsz * Lseq)      // 8192

typedef __attribute__((ext_vector_type(16))) _Float16 v16h;
typedef __attribute__((ext_vector_type(8)))  float    v8f;

union Frag { v16h v; uint32_t u[8]; };

__device__ __forceinline__ uint32_t pack2h(float a, float b) {
  union { _Float16 h[2]; uint32_t u; } t;
  t.h[0] = (_Float16)a; t.h[1] = (_Float16)b;
  return t.u;
}

__device__ __forceinline__ v8f wmma_f16(v16h a, v16h b, v8f c) {
  return __builtin_amdgcn_wmma_f32_16x16x32_f16(
      /*neg_a=*/false, a, /*neg_b=*/false, b,
      /*c_mod=*/(short)0, c, /*reuse_a=*/false, /*reuse_b=*/false);
}

// A-fragment K index for VGPR pair vv (16-bit A 16x32, matches B):
// lanes 0-15: K 0..15 contiguous pairs
// lanes 16-31: +16
__device__ __forceinline__ int a_kpair(int vv, int hi) {
  return 2 * vv + (hi ? 16 : 0);
}

__device__ __forceinline__ Frag load_a_frag(const _Float16* rowp, int hi) {
  Frag f;
  const uint32_t* p = (const uint32_t*)rowp;
  #pragma unroll
  for (int vv = 0; vv < 8; ++vv) f.u[vv] = p[a_kpair(vv, hi) >> 1];
  return f;
}

// B-fragment (16-bit B 32x16): col = lane&15; lanes 0-15 hold K 0..15,
// lanes 16-31 hold K 16..31, pairs contiguous.
__device__ __forceinline__ Frag load_b_frag(const _Float16* colp, int hi) {
  Frag f;
  const uint32_t* p = (const uint32_t*)colp;
  #pragma unroll
  for (int vv = 0; vv < 8; ++vv) f.u[vv] = p[vv + 8 * hi];
  return f;
}

// ---------------- conv1d embedding (wrap pad) ----------------
__global__ void conv_emb_kernel(const float* __restrict__ x,
                                const float* __restrict__ ker,
                                float* __restrict__ h,
                                float* __restrict__ emb_out) {
  int idx = blockIdx.x * 256 + threadIdx.x;     // B*L*D threads
  int d = idx & (Dm - 1);
  int l = (idx >> 9) & (Lseq - 1);
  int b = idx >> 18;
  float acc = 0.f;
  #pragma unroll
  for (int t = 0; t < 3; ++t) {
    int ls = (l - 1 + t + Lseq) & (Lseq - 1);   // wrap
    const float* xp = x + ((size_t)b * Lseq + ls) * Cin;
    const float* kp = ker + (size_t)d * (Cin * 3) + t;
    #pragma unroll
    for (int c = 0; c < Cin; ++c) acc += xp[c] * kp[c * 3];
  }
  h[idx] = acc;
  emb_out[idx] = acc;
}

// ---------------- generic WMMA GEMM: C = act(A @ B + bias) ----------------
// A: M x K (row stride lda), B: K x N (row stride ldb), C: M x N (row stride ldc)
// block = 256 threads = 8 waves; block tile 128(M) x 64(N), K step 32.
// Each wave owns a 32x32 sub-block -> 4 wmma per K-step.
__global__ __launch_bounds__(256)
void gemm_kernel(const float* __restrict__ A, int lda,
                 const float* __restrict__ Bm, int ldb,
                 const float* __restrict__ bias,
                 float* __restrict__ C, int ldc,
                 int M, int N, int K, int act) {
  __shared__ alignas(16) _Float16 sA[128 * 32];       // [m][k]
  __shared__ alignas(16) _Float16 sB[64 * 32];        // [n][k] (transposed stage)
  const int tid = threadIdx.x;
  const int lane = tid & 31;
  const int w = tid >> 5;
  const int m16 = lane & 15;
  const int hi = lane >> 4;
  const int bm = blockIdx.y * 128;
  const int bn = blockIdx.x * 64;
  const int wm = (w >> 1) * 32;                       // wave row base in tile
  const int wn = (w & 1) * 32;                        // wave col base in tile

  v8f acc00 = {}, acc01 = {}, acc10 = {}, acc11 = {};
  for (int k0 = 0; k0 < K; k0 += 32) {
    { // stage A (128x32): 16 consecutive floats/thread, packed b32 LDS stores
      int row = tid >> 1, c0 = (tid & 1) * 16;
      const float* ap = A + (size_t)(bm + row) * lda + k0 + c0;
      uint32_t* dst = (uint32_t*)&sA[row * 32 + c0];
      #pragma unroll
      for (int j = 0; j < 8; ++j) dst[j] = pack2h(ap[2 * j], ap[2 * j + 1]);
      if (k0 + 32 < K) __builtin_prefetch(ap + 32, 0, 1);
    }
    { // stage B transposed (32x64 -> [n][k]), coalesced along N
      int kr = tid >> 3, n0 = (tid & 7) * 8;
      const float* bp = Bm + (size_t)(k0 + kr) * ldb + bn + n0;
      #pragma unroll
      for (int j = 0; j < 8; ++j) sB[(n0 + j) * 32 + kr] = (_Float16)bp[j];
      if (k0 + 32 < K) __builtin_prefetch(bp + (size_t)32 * ldb, 0, 1);
    }
    __syncthreads();
    Frag a0 = load_a_frag(&sA[(wm + m16) * 32], hi);
    Frag a1 = load_a_frag(&sA[(wm + 16 + m16) * 32], hi);
    Frag b0 = load_b_frag(&sB[(wn + m16) * 32], hi);
    Frag b1 = load_b_frag(&sB[(wn + 16 + m16) * 32], hi);
    acc00 = wmma_f16(a0.v, b0.v, acc00);
    acc01 = wmma_f16(a0.v, b1.v, acc01);
    acc10 = wmma_f16(a1.v, b0.v, acc10);
    acc11 = wmma_f16(a1.v, b1.v, acc11);
    __syncthreads();
  }
  // epilogue: bias (+ optional exact GELU)
  #pragma unroll
  for (int i = 0; i < 2; ++i) {
    #pragma unroll
    for (int j = 0; j < 2; ++j) {
      v8f acc = i ? (j ? acc11 : acc10) : (j ? acc01 : acc00);
      int col = bn + wn + j * 16 + m16;
      float bv = bias[col];
      #pragma unroll
      for (int r = 0; r < 8; ++r) {
        int row = bm + wm + i * 16 + r + 8 * hi;
        float val = acc[r] + bv;
        if (act == 1) val = 0.5f * val * (1.0f + erff(val * 0.70710678118654752f));
        C[(size_t)row * ldc + col] = val;
      }
    }
  }
}

// ---------------- fused flash attention ----------------
// One wave handles (b, head, 16-row query tile); streams K/V in 32-col chunks
// with online softmax. Scores never touch global memory.
__global__ __launch_bounds__(128)
void flash_attn_kernel(const float* __restrict__ q,
                       const float* __restrict__ k,
                       const float* __restrict__ v,
                       float* __restrict__ o) {
  __shared__ alignas(16) _Float16 lds_p[4][16 * 32];  // P tile staging (per wave)
  __shared__ float lds_m[4][16], lds_l[4][16], lds_c[4][16], lds_a[4][16];
  const int w = threadIdx.x >> 5;
  const int lane = threadIdx.x & 31;
  const int m16 = lane & 15;
  const int hi = lane >> 4;
  const int widx = blockIdx.x * 4 + w;
  const int lt = widx & 31;              // L/16 tiles
  const int hh = (widx >> 5) & (Hh - 1);
  const int bb = widx >> 8;

  const size_t headbase = ((size_t)bb * Lseq) * Dm + (size_t)hh * Eh;

  // Q as two A-fragments (E = 64 = 2 x K32), loaded once
  Frag qa[2];
  {
    const float* qp = q + headbase + (size_t)(lt * 16 + m16) * Dm;
    #pragma unroll
    for (int e0 = 0; e0 < 2; ++e0)
      #pragma unroll
      for (int vv = 0; vv < 8; ++vv) {
        int kk = a_kpair(vv, hi) + e0 * 32;
        qa[e0].u[vv] = pack2h(qp[kk], qp[kk + 1]);
      }
  }
  v8f oacc[4] = {{}, {}, {}, {}};
  if (lane < 16) { lds_m[w][lane] = -1.0e30f; lds_l[w][lane] = 0.f; }
  __syncthreads();

  for (int s0 = 0; s0 < Lseq; s0 += 32) {
    // ---- S = (Q @ K^T) * scale, two 16x16 tiles ----
    v8f sf0 = {}, sf1 = {};
    #pragma unroll
    for (int e0 = 0; e0 < 2; ++e0) {
      Frag kb0, kb1;
      const float* kp0 = k + headbase + (size_t)(s0 + m16) * Dm + e0 * 32;
      const float* kp1 = k + headbase + (size_t)(s0 + 16 + m16) * Dm + e0 * 32;
      #pragma unroll
      for (int vv = 0; vv < 8; ++vv) {
        int kk = 2 * vv + (hi ? 16 : 0);
        kb0.u[vv] = pack2h(kp0[kk], kp0[kk + 1]);
        kb1.u[vv] = pack2h(kp1[kk], kp1[kk + 1]);
      }
      sf0 = wmma_f16(qa[e0].v, kb0.v, sf0);
      sf1 = wmma_f16(qa[e0].v, kb1.v, sf1);
    }
    #pragma unroll
    for (int r = 0; r < 8; ++r) { sf0[r] *= 0.125f; sf1[r] *= 0.125f; }

    // ---- chunk row-max (reduce across 16-lane half-groups) ----
    #pragma unroll
    for (int r = 0; r < 8; ++r) {
      float t = fmaxf(sf0[r], sf1[r]);
      t = fmaxf(t, __shfl_xor(t, 1, 32));
      t = fmaxf(t, __shfl_xor(t, 2, 32));
      t = fmaxf(t, __shfl_xor(t, 4, 32));
      t = fmaxf(t, __shfl_xor(t, 8, 32));
      if (m16 == 0) lds_c[w][r + 8 * hi] = t;
    }
    __syncthreads();
    // ---- P = exp(S - m_new) ----
    #pragma unroll
    for (int r = 0; r < 8; ++r) {
      int row = r + 8 * hi;
      float mn = fmaxf(lds_m[w][row], lds_c[w][row]);
      sf0[r] = __expf(sf0[r] - mn);
      sf1[r] = __expf(sf1[r] - mn);
    }
    // ---- chunk row-sum + running stats update ----
    #pragma unroll
    for (int r = 0; r < 8; ++r) {
      float t = sf0[r] + sf1[r];
      t += __shfl_xor(t, 1, 32);
      t += __shfl_xor(t, 2, 32);
      t += __shfl_xor(t, 4, 32);
      t += __shfl_xor(t, 8, 32);
      if (m16 == 0) {
        int row = r + 8 * hi;
        float mo = lds_m[w][row];
        float mn = fmaxf(mo, lds_c[w][row]);
        float alpha = __expf(mo - mn);
        lds_l[w][row] = lds_l[w][row] * alpha + t;
        lds_m[w][row] = mn;
        lds_a[w][row] = alpha;
      }
    }
    __syncthreads();
    // ---- rescale O by alpha ----
    #pragma unroll
    for (int nt = 0; nt < 4; ++nt)
      #pragma unroll
      for (int r = 0; r < 8; ++r) oacc[nt][r] *= lds_a[w][r + 8 * hi];

    // ---- C-layout P -> A-layout via LDS ----
    #pragma unroll
    for (int r = 0; r < 8; ++r) {
      int row = r + 8 * hi;
      lds_p[w][row * 32 + m16]      = (_Float16)sf0[r];
      lds_p[w][row * 32 + 16 + m16] = (_Float16)sf1[r];
    }
    __syncthreads();
    Frag pa = load_a_frag(&lds_p[w][m16 * 32], hi);

    // ---- O += P @ V (4 N-tiles over E=64) ----
    #pragma unroll
    for (int nt = 0; nt < 4; ++nt) {
      Frag vb;
      #pragma unroll
      for (int vv = 0; vv < 8; ++vv) {
        int kk = 2 * vv + (hi ? 16 : 0);
        const float* vp = v + headbase + (size_t)(s0 + kk) * Dm + nt * 16 + m16;
        vb.u[vv] = pack2h(vp[0], vp[Dm]);
      }
      oacc[nt] = wmma_f16(pa.v, vb.v, oacc[nt]);
    }
    __syncthreads();
  }
  // ---- normalize and write context ----
  #pragma unroll
  for (int nt = 0; nt < 4; ++nt)
    #pragma unroll
    for (int r = 0; r < 8; ++r) {
      int row = r + 8 * hi;
      float val = oacc[nt][r] / lds_l[w][row];
      o[headbase + (size_t)(lt * 16 + row) * Dm + nt * 16 + m16] = val;
    }
}

// ---------------- residual-add + LayerNorm (D=512) ----------------
__global__ __launch_bounds__(128)
void add_ln_kernel(const float* __restrict__ a, const float* __restrict__ res,
                   const float* __restrict__ g, const float* __restrict__ beta,
                   float* __restrict__ out, int hasRes) {
  __shared__ float red[128];
  const int row = blockIdx.x, tid = threadIdx.x;
  const float* pa = a + (size_t)row * Dm;
  const float* pr = res ? res + (size_t)row * Dm : nullptr;
  float vals[4]; float s = 0.f;
  #pragma unroll
  for (int j = 0; j < 4; ++j) {
    int c = tid + j * 128;
    float x = pa[c];
    if (hasRes) x += pr[c];
    vals[j] = x; s += x;
  }
  red[tid] = s; __syncthreads();
  for (int st = 64; st > 0; st >>= 1) { if (tid < st) red[tid] += red[tid + st]; __syncthreads(); }
  float mean = red[0] * (1.0f / Dm);
  __syncthreads();
  float s2 = 0.f;
  #pragma unroll
  for (int j = 0; j < 4; ++j) { float d = vals[j] - mean; s2 += d * d; }
  red[tid] = s2; __syncthreads();
  for (int st = 64; st > 0; st >>= 1) { if (tid < st) red[tid] += red[tid + st]; __syncthreads(); }
  float inv = rsqrtf(red[0] * (1.0f / Dm) + 1e-5f);
  __syncthreads();
  #pragma unroll
  for (int j = 0; j < 4; ++j) {
    int c = tid + j * 128;
    out[(size_t)row * Dm + c] = (vals[j] - mean) * inv * g[c] + beta[c];
  }
}

// ---------------- enc_res slice copy ----------------
__global__ void copy_res_kernel(const float* __restrict__ enc, float* __restrict__ out) {
  int idx = blockIdx.x * 256 + threadIdx.x;    // B*L*384 threads
  int c = idx % (Dm - DRESm);
  int row = idx / (Dm - DRESm);
  out[idx] = enc[(size_t)row * Dm + c];
}

// ---------------- host orchestration ----------------
static inline void launch_gemm(const float* A, int lda, const float* B, int ldb,
                               const float* bias, float* C, int ldc,
                               int M, int N, int K, int act, hipStream_t s) {
  dim3 grid(N / 64, M / 128);
  gemm_kernel<<<grid, 256, 0, s>>>(A, lda, B, ldb, bias, C, ldc, M, N, K, act);
}

extern "C" void kernel_launch(void* const* d_in, const int* in_sizes, int n_in,
                              void* d_out, int out_size, void* d_ws, size_t ws_size,
                              hipStream_t stream) {
  const float* x     = (const float*)d_in[0];
  const float* ker   = (const float*)d_in[1];
  const float* Wq    = (const float*)d_in[2];
  const float* bq    = (const float*)d_in[3];
  const float* Wk    = (const float*)d_in[4];
  const float* bk    = (const float*)d_in[5];
  const float* Wv    = (const float*)d_in[6];
  const float* bv    = (const float*)d_in[7];
  // d_in[8] Wsig, d_in[9] bsig: dead code in the reference (prior is deleted)
  const float* Wo    = (const float*)d_in[10];
  const float* bo    = (const float*)d_in[11];
  const float* W1    = (const float*)d_in[12];
  const float* b1    = (const float*)d_in[13];
  const float* W2    = (const float*)d_in[14];
  const float* b2    = (const float*)d_in[15];
  const float* n1g   = (const float*)d_in[16];
  const float* n1b   = (const float*)d_in[17];
  const float* n2g   = (const float*)d_in[18];
  const float* n2b   = (const float*)d_in[19];
  const float* nfg   = (const float*)d_in[20];
  const float* nfb   = (const float*)d_in[21];
  const float* Wproj = (const float*)d_in[22];
  const float* bproj = (const float*)d_in[23];
  const float* Wfea  = (const float*)d_in[24];
  const float* bfea  = (const float*)d_in[25];

  float* out = (float*)d_out;
  float* ws  = (float*)d_ws;

  const size_t S = (size_t)Mrows * Dm;              // 4,194,304 floats per slab
  float* hbuf = ws;                                  // h
  float* qbuf = ws + S;                              // q, then attn_out
  float* kbuf = ws + 2 * S;                          // k, then y
  float* vbuf = ws + 3 * S;                          // v, then x1
  float* cbuf = ws + 4 * S;                          // attention context, then enc
  float* fbuf = ws + 5 * S;                          // FFN hidden (4 slabs)

  float* fea  = out;                                 // (B,L,512)
  float* res  = out + S;                             // (B,L,384)
  float* rec  = out + S + (size_t)Mrows * (Dm - DRESm);
  float* embo = rec + S;                             // (B,L,512)

  conv_emb_kernel<<<(Mrows * Dm) / 256, 256, 0, stream>>>(x, ker, hbuf, embo);

  for (int l = 0; l < NLm; ++l) {
    const size_t wDD = (size_t)l * Dm * Dm;
    launch_gemm(hbuf, Dm, Wq + wDD, Dm, bq + l * Dm, qbuf, Dm, Mrows, Dm, Dm, 0, stream);
    launch_gemm(hbuf, Dm, Wk + wDD, Dm, bk + l * Dm, kbuf, Dm, Mrows, Dm, Dm, 0, stream);
    launch_gemm(hbuf, Dm, Wv + wDD, Dm, bv + l * Dm, vbuf, Dm, Mrows, Dm, Dm, 0, stream);

    // fused attention: q,k,v -> context (cbuf)
    flash_attn_kernel<<<(Bsz * Hh * (Lseq / 16)) / 4, 128, 0, stream>>>(qbuf, kbuf, vbuf, cbuf);

    // attn_out = ctx @ Wo + bo   (q slab is dead -> reuse)
    launch_gemm(cbuf, Dm, Wo + wDD, Dm, bo + l * Dm, qbuf, Dm, Mrows, Dm, Dm, 0, stream);
    // x1 = LN(h + attn_out)      (v slab dead -> reuse)
    add_ln_kernel<<<Mrows, 128, 0, stream>>>(hbuf, qbuf, n1g + l * Dm, n1b + l * Dm, vbuf, 1);
    // ffn hidden = gelu(x1 @ W1 + b1)
    launch_gemm(vbuf, Dm, W1 + (size_t)l * Dm * DFFm, DFFm, b1 + l * DFFm,
                fbuf, DFFm, Mrows, DFFm, Dm, 1, stream);
    // y = hidden @ W2 + b2       (k slab dead -> reuse)
    launch_gemm(fbuf, DFFm, W2 + (size_t)l * DFFm * Dm, Dm, b2 + l * Dm,
                kbuf, Dm, Mrows, Dm, DFFm, 0, stream);
    // h = LN(x1 + y)
    add_ln_kernel<<<Mrows, 128, 0, stream>>>(vbuf, kbuf, n2g + l * Dm, n2b + l * Dm, hbuf, 1);
  }

  // enc = LN(h)  (ctx slab dead -> reuse)
  add_ln_kernel<<<Mrows, 128, 0, stream>>>(hbuf, nullptr, nfg, nfb, cbuf, 0);
  // enc_res = enc[:, :, :384]
  copy_res_kernel<<<(Mrows * (Dm - DRESm)) / 256, 256, 0, stream>>>(cbuf, res);
  // rec_temp = enc @ Wproj + bproj
  launch_gemm(cbuf, Dm, Wproj, Dm, bproj, rec, Dm, Mrows, Dm, Dm, 0, stream);
  // fea_t = enc[:, :, 384:] @ Wfea + bfea   (A offset 384, lda=512, K=128)
  launch_gemm(cbuf + (Dm - DRESm), Dm, Wfea, Dm, bfea, fea, Dm, Mrows, Dm, DRESm, 0, stream);
}
